// MLC_65094524338971
// MI455X (gfx1250) — matrix operand order repaired and checked by reading, not compiled, not run on hardware
//
#include <hip/hip_runtime.h>
#include <hip/hip_bf16.h>

// Sizes from the reference
#define NROWS   65536
#define DVIS    1024
#define CLASSES 210
#define CPAD    224      // 14 x 16 class tiles
#define NTILES  14
#define EMBED   512
#define TOPK    10

typedef __attribute__((ext_vector_type(8)))  float        v8f;
typedef __attribute__((ext_vector_type(8)))  unsigned int v8u;
typedef __attribute__((ext_vector_type(16))) __bf16       v16bf;

// LDS layout (dynamic shared memory)
#define SW_STRIDE   40                        // shorts per class row (32 + 8 pad) -> 80B, conflict-free
#define SW_BYTES    (CPAD * SW_STRIDE * 2)    // 17920 per buffer (x2, double buffered)
#define SL_STRIDE   226                       // floats per logit row (224 + 2 pad)
#define SL_BYTES    (128 * SL_STRIDE * 4)     // 115712
#define SB_BYTES    (CPAD * 4)                // 896
#define SMEM_BYTES  (2 * SW_BYTES + SL_BYTES + SB_BYTES)  // 152448

__device__ __forceinline__ unsigned pack_bf16(float lo, float hi) {
#if __has_builtin(__builtin_amdgcn_cvt_pk_bf16_f32)
    typedef __attribute__((ext_vector_type(2))) __bf16 v2bf;
    v2bf p = __builtin_amdgcn_cvt_pk_bf16_f32(lo, hi);
    return __builtin_bit_cast(unsigned, p);
#else
    unsigned a = __builtin_bit_cast(unsigned, lo);
    unsigned b = __builtin_bit_cast(unsigned, hi);
    a = (a + 0x7FFFu + ((a >> 16) & 1u)) >> 16;   // round-to-nearest-even
    b = (b + 0x7FFFu + ((b >> 16) & 1u)) >> 16;
    return a | (b << 16);
#endif
}

// ---------------------------------------------------------------------------
// Kernel 1: convert W (fp32, 210x1024) -> bf16 padded to 224x1024; pad bias.
// ---------------------------------------------------------------------------
__global__ void MLC_prep_kernel(const float* __restrict__ W,
                                const float* __restrict__ b,
                                unsigned short* __restrict__ Wb,
                                float* __restrict__ biasPad) {
    int i = blockIdx.x * 256 + threadIdx.x;          // over CPAD*DVIS
    int c = i >> 10;
    int k = i & 1023;
    float v = (c < CLASSES) ? W[c * DVIS + k] : 0.0f;
    unsigned u = __builtin_bit_cast(unsigned, v);
    u = (u + 0x7FFFu + ((u >> 16) & 1u)) >> 16;
    Wb[i] = (unsigned short)u;
    if (i < CPAD) biasPad[i] = (i < CLASSES) ? b[i] : -3.0e38f;
}

// ---------------------------------------------------------------------------
// Kernel 2: fused bf16-WMMA GEMM (X @ Wb^T + bias) + per-row top-10 selection.
// 256 threads = 8 wave32; each wave computes a 16x224 logit tile via
// v_wmma_f32_16x16x32_bf16.
//   - W k-slices double-buffered in LDS, filled with
//     GLOBAL_LOAD_ASYNC_TO_LDS_B128 (ASYNCcnt), overlapped with compute.
//   - B fragments: depth-2 pipeline in two *distinct* register blocks so
//     each WMMA waits dscnt<=2, not 0.
//   - A operand (X, fp32->bf16) pipelined one k-iteration ahead.
// ---------------------------------------------------------------------------
__global__ __launch_bounds__(256)
void MLC_gemm_topk_kernel(const float* __restrict__ X,
                          const unsigned short* __restrict__ Wb,
                          const float* __restrict__ biasPad,
                          int* __restrict__ outIdx) {
    extern __shared__ char smem[];
    unsigned short* sW      = (unsigned short*)smem;              // 2 x [CPAD][SW_STRIDE] bf16
    float*          sLogits = (float*)(smem + 2 * SW_BYTES);      // [128][SL_STRIDE]
    float*          sBias   = (float*)(smem + 2 * SW_BYTES + SL_BYTES);

    const int tid  = threadIdx.x;
    const int wave = tid >> 5;
    const int lane = tid & 31;
    const int half = lane >> 4;       // 0: lanes 0-15, 1: lanes 16-31
    const int l16  = lane & 15;

    if (tid < CPAD) sBias[tid] = biasPad[tid];

    const long rowBase = (long)blockIdx.x * 128;
    const float* xrow = X + (rowBase + wave * 16 + l16) * DVIS;

    v8f acc[NTILES] = {};

    // Async-fill one W k-slice (224 classes x 32 k bf16 = 896 x b128) into LDS
    // buffer `buf` using the CDNA5 async-to-LDS path (no VGPR round trip).
    auto stage_async = [&](int kb, int buf) {
        for (int i = tid; i < CPAD * 4; i += 256) {
            int c  = i >> 2;
            int ch = i & 3;
            unsigned lds_off = (unsigned)(buf * SW_BYTES + c * (SW_STRIDE * 2) + ch * 16);
            unsigned goff    = (unsigned)(c * (DVIS * 2) + kb * 2 + ch * 16);
            asm volatile("global_load_async_to_lds_b128 %0, %1, %2"
                         :
                         : "v"(lds_off), "v"(goff), "s"(Wb)
                         : "memory");
        }
    };

    // A-operand raw loads for one k-step (ISA A layout:
    // lanes 0-15: K 0..7 / 16..23; lanes 16-31: K 8..15 / 24..31)
    auto loadA = [&](int kb, float4* r) {
        r[0] = *(const float4*)(xrow + kb + half * 8);
        r[1] = *(const float4*)(xrow + kb + half * 8 + 4);
        r[2] = *(const float4*)(xrow + kb + 16 + half * 8);
        r[3] = *(const float4*)(xrow + kb + 16 + half * 8 + 4);
    };

    stage_async(0, 0);
    float4 ra[4];
    loadA(0, ra);
    asm volatile("s_wait_asynccnt 0x0" ::: "memory");
    __syncthreads();

    for (int it = 0; it < DVIS / 32; ++it) {
        const int kb = it * 32;
        // Kick off next slice's async LDS fill and next A loads first so their
        // latency hides behind this iteration's 14 WMMAs.
        if (it + 1 < DVIS / 32) stage_async(kb + 32, (it + 1) & 1);
        float4 rn[4] = {ra[0], ra[1], ra[2], ra[3]};
        if (it + 1 < DVIS / 32) loadA(kb + 32, rn);
        if (kb + 96 < DVIS) __builtin_prefetch(xrow + kb + 96, 0, 1);

        const unsigned short* sWc = sW + (it & 1) * (SW_BYTES / 2);

        // Pack current A fragment (loaded last iteration) to bf16.
        v8u ap;
        ap[0] = pack_bf16(ra[0].x, ra[0].y); ap[1] = pack_bf16(ra[0].z, ra[0].w);
        ap[2] = pack_bf16(ra[1].x, ra[1].y); ap[3] = pack_bf16(ra[1].z, ra[1].w);
        ap[4] = pack_bf16(ra[2].x, ra[2].y); ap[5] = pack_bf16(ra[2].z, ra[2].w);
        ap[6] = pack_bf16(ra[3].x, ra[3].y); ap[7] = pack_bf16(ra[3].z, ra[3].w);
        v16bf afrag = __builtin_bit_cast(v16bf, ap);

        // B fragment loader: 32x16 bf16; lane -> column, K 8h..8h+7 / 16+8h..23+8h per half
        auto loadB = [&](int n) -> v16bf {
            int c = n * 16 + l16;
            const uint4* bp = (const uint4*)(sWc + c * SW_STRIDE + half * 8);
            uint4 x0 = bp[0];
            uint4 x1 = bp[2];
            v8u bpk;
            bpk[0] = x0.x; bpk[1] = x0.y; bpk[2] = x0.z; bpk[3] = x0.w;
            bpk[4] = x1.x; bpk[5] = x1.y; bpk[6] = x1.z; bpk[7] = x1.w;
            return __builtin_bit_cast(v16bf, bpk);
        };

        // Depth-2 pipeline in two distinct live registers (non-coalescable):
        // each WMMA only waits for its own fragment; the refill of the other
        // buffer stays outstanding (s_wait_dscnt <= 2).
        v16bf b0 = loadB(0);
        v16bf b1 = loadB(1);
        #pragma unroll
        for (int n = 0; n < NTILES; n += 2) {
            acc[n] = __builtin_amdgcn_wmma_f32_16x16x32_bf16(
                false, afrag, false, b0, (short)0, acc[n], false, false);
            if (n + 2 < NTILES) b0 = loadB(n + 2);
            acc[n + 1] = __builtin_amdgcn_wmma_f32_16x16x32_bf16(
                false, afrag, false, b1, (short)0, acc[n + 1], false, false);
            if (n + 3 < NTILES) b1 = loadB(n + 3);
        }

        ra[0] = rn[0]; ra[1] = rn[1]; ra[2] = rn[2]; ra[3] = rn[3];

        // Next slice fully in LDS (per-wave ASYNCcnt) before the barrier.
        asm volatile("s_wait_asynccnt 0x0" ::: "memory");
        __syncthreads();
    }

    // Spill logits (+bias) to LDS. C/D layout: VGPR r -> M = r + 8*half, N = l16.
    #pragma unroll
    for (int n = 0; n < NTILES; ++n) {
        int col = n * 16 + l16;
        float bb = sBias[col];
        #pragma unroll
        for (int r = 0; r < 8; ++r) {
            int lrow = wave * 16 + half * 8 + r;
            sLogits[lrow * SL_STRIDE + col] = acc[n][r] + bb;
        }
    }
    __syncthreads();

    // Top-10 selection per row (ties -> lowest index, matching lax.top_k).
    // float2 LDS reads: row stride 904B is 8B-aligned, CLASSES is even.
    if (tid < 128) {
        float* rowp = sLogits + tid * SL_STRIDE;
        long gr = rowBase + tid;
        for (int kk = 0; kk < TOPK; ++kk) {
            float best = -3.0e38f;
            int   bi   = 0;
            #pragma unroll 5
            for (int c = 0; c < CLASSES; c += 2) {
                float2 v = *(const float2*)(rowp + c);
                if (v.x > best) { best = v.x; bi = c; }
                if (v.y > best) { best = v.y; bi = c + 1; }
            }
            rowp[bi] = -3.0e38f;
            outIdx[gr * TOPK + kk] = bi;
        }
    }
}

// ---------------------------------------------------------------------------
// Kernel 3: embeddings[row] = sum_k emb_table[topk[row][k]]  (table is L2-hot)
// ---------------------------------------------------------------------------
__global__ __launch_bounds__(256)
void MLC_embed_kernel(const int* __restrict__ outIdx,
                      const float* __restrict__ emb,
                      float* __restrict__ outEmb) {
    long row = blockIdx.x;
    int  t   = threadIdx.x;
    int id[TOPK];
    #pragma unroll
    for (int k = 0; k < TOPK; ++k) id[k] = outIdx[row * TOPK + k];
    float a0 = 0.0f, a1 = 0.0f;
    #pragma unroll
    for (int k = 0; k < TOPK; ++k) {
        const float* e = emb + (long)id[k] * EMBED;
        a0 += e[t];
        a1 += e[t + 256];
    }
    outEmb[row * EMBED + t]       = a0;
    outEmb[row * EMBED + t + 256] = a1;
}

// ---------------------------------------------------------------------------
extern "C" void kernel_launch(void* const* d_in, const int* in_sizes, int n_in,
                              void* d_out, int out_size, void* d_ws, size_t ws_size,
                              hipStream_t stream) {
    const float* X   = (const float*)d_in[0];   // (65536, 1024) fp32
    const float* W   = (const float*)d_in[1];   // (210, 1024)   fp32
    const float* b   = (const float*)d_in[2];   // (210,)        fp32
    const float* emb = (const float*)d_in[3];   // (210, 512)    fp32

    unsigned short* Wb      = (unsigned short*)d_ws;                    // 224*1024 bf16
    float*          biasPad = (float*)((char*)d_ws + CPAD * DVIS * 2);  // 224 fp32

    int*   outIdx = (int*)d_out;                        // N*K int32 (bit-stored)
    float* outEmb = (float*)d_out + (long)NROWS * TOPK; // N*EMBED fp32

    MLC_prep_kernel<<<(CPAD * DVIS) / 256, 256, 0, stream>>>(W, b, Wb, biasPad);
    MLC_gemm_topk_kernel<<<NROWS / 128, 256, SMEM_BYTES, stream>>>(X, Wb, biasPad, outIdx);
    MLC_embed_kernel<<<NROWS, 256, 0, stream>>>(outIdx, emb, outEmb);
}
